// LlamaAttention_35055523070352
// MI455X (gfx1250) — matrix operand order repaired and checked by reading, not compiled, not run on hardware
//
#include <hip/hip_runtime.h>
#include <hip/hip_bf16.h>

// ---------------- problem constants ----------------
constexpr int T_   = 4096;
constexpr int D_   = 2048;
constexpr int NH_  = 16;
constexpr int HD_  = 128;           // head dim
constexpr float SCALE_ = 0.08838834764831845f;  // 1/sqrt(128)

// ---------------- types ----------------
typedef __bf16 bf16_t;
typedef bf16_t bf16x16 __attribute__((ext_vector_type(16)));
typedef bf16_t bf16x8  __attribute__((ext_vector_type(8)));
typedef float  v8f     __attribute__((ext_vector_type(8)));
typedef int    v4i     __attribute__((ext_vector_type(4)));

// ---------------- CDNA5 async global->LDS (guarded; ds_store fallback) ------
#if defined(__has_builtin)
#if __has_builtin(__builtin_amdgcn_global_load_async_to_lds_b128)
#define USE_ASYNC_LDS 1
#endif
#endif
#ifndef USE_ASYNC_LDS
#define USE_ASYNC_LDS 0
#endif

// copy 16 bytes (8 bf16) global -> LDS, per lane
static __device__ __forceinline__ void copy_b128(const bf16_t* g, bf16_t* l) {
#if USE_ASYNC_LDS
  __builtin_amdgcn_global_load_async_to_lds_b128(
      (__attribute__((address_space(1))) v4i*)(unsigned long long)g,
      (__attribute__((address_space(3))) v4i*)(unsigned)(unsigned long long)l,
      0, 0);
#else
  *(bf16x8*)l = *(const bf16x8*)g;
#endif
}

static __device__ __forceinline__ void async_wait_all() {
#if USE_ASYNC_LDS
#if __has_builtin(__builtin_amdgcn_s_wait_asynccnt)
  __builtin_amdgcn_s_wait_asynccnt(0);
#else
  asm volatile("s_wait_asynccnt 0x0" ::: "memory");
#endif
#endif
}

// ---------------- helpers ----------------
static __device__ __forceinline__ bf16_t to_bf16(float f) {
  unsigned u = __builtin_bit_cast(unsigned, f);
  u += 0x7FFFu + ((u >> 16) & 1u);                 // round-to-nearest-even
  unsigned short h = (unsigned short)(u >> 16);
  return __builtin_bit_cast(bf16_t, h);
}
static __device__ __forceinline__ float bf16_to_f32(bf16_t b) {
  unsigned short h = __builtin_bit_cast(unsigned short, b);
  unsigned u = ((unsigned)h) << 16;
  return __builtin_bit_cast(float, u);
}

// A-fragment (16x32 bf16, row-major source): lane row = lane&15,
// K = (lane>>4)*16 + 0..15 contiguous (matches load_bfrag).
static __device__ __forceinline__ bf16x16 load_afrag(const bf16_t* p, int hw) {
  bf16x16 r;
  bf16x8 lo = *(const bf16x8*)(p + hw * 16);
  bf16x8 hi = *(const bf16x8*)(p + hw * 16 + 8);
#pragma unroll
  for (int j = 0; j < 8; ++j) { r[j] = lo[j]; r[j + 8] = hi[j]; }
  return r;
}

// B-fragment (32x16 bf16): lane column = lane&15, K = (lane>>4)*16 + 0..15,
// 16 contiguous elements (caller applies the (lane>>4)*16 offset).
static __device__ __forceinline__ bf16x16 load_bfrag(const bf16_t* p) {
  bf16x16 r;
  bf16x8 lo = *(const bf16x8*)(p);
  bf16x8 hi = *(const bf16x8*)(p + 8);
#pragma unroll
  for (int j = 0; j < 8; ++j) { r[j] = lo[j]; r[j + 8] = hi[j]; }
  return r;
}

static __device__ __forceinline__ void store_out(float* p, float v)  { *p = v; }
static __device__ __forceinline__ void store_out(bf16_t* p, float v) { *p = to_bf16(v); }

// ---------------------------------------------------------------------------
// fp32 -> bf16 streaming convert (8 elements / thread, b128 in, b128 out)
// ---------------------------------------------------------------------------
__global__ __launch_bounds__(256) void cvt_bf16_kernel(
    const float* __restrict__ src, bf16_t* __restrict__ dst) {
  const size_t base = ((size_t)blockIdx.x * 256 + threadIdx.x) * 8;
  float4 f0 = *(const float4*)(src + base);
  float4 f1 = *(const float4*)(src + base + 4);
  bf16x8 r;
  r[0] = to_bf16(f0.x); r[1] = to_bf16(f0.y);
  r[2] = to_bf16(f0.z); r[3] = to_bf16(f0.w);
  r[4] = to_bf16(f1.x); r[5] = to_bf16(f1.y);
  r[6] = to_bf16(f1.z); r[7] = to_bf16(f1.w);
  *(bf16x8*)(dst + base) = r;
}

// ---------------------------------------------------------------------------
// GEMM: out[m, n] = sum_k A[m, k] * B[n, k]   (both bf16, row-major, B^T form)
// One wave computes a 32x64 tile: 2 A frags x 4 B frags = 8 WMMAs per K-step.
// ---------------------------------------------------------------------------
template <typename OT>
__global__ __launch_bounds__(256) void gemm_bt_bf16_kernel(
    const bf16_t* __restrict__ A, const bf16_t* __restrict__ B,
    OT* __restrict__ out, int M, int N, int K) {
  const int lane = threadIdx.x & 31;
  const int wave = threadIdx.x >> 5;
  const int nstrips = N >> 6;                       // N / 64
  const int id = blockIdx.x * 8 + wave;
  const int mt = id / nstrips;                      // 32-row tile
  const int nb = id % nstrips;
  if (mt * 32 >= M) return;

  const int m  = lane & 15;
  const int hw = lane >> 4;

  v8f acc[2][4];
#pragma unroll
  for (int r = 0; r < 2; ++r)
#pragma unroll
    for (int s = 0; s < 4; ++s) acc[r][s] = v8f{0,0,0,0,0,0,0,0};

  const bf16_t* a0p = A + (size_t)(mt * 32 + m) * K;
  const bf16_t* a1p = A + (size_t)(mt * 32 + 16 + m) * K;
  const bf16_t* bp  = B + (size_t)(nb * 64 + m) * K + hw * 16;

  for (int k0 = 0; k0 < K; k0 += 32) {
    bf16x16 a0 = load_afrag(a0p + k0, hw);
    bf16x16 a1 = load_afrag(a1p + k0, hw);
    if (k0 + 32 < K) {
      __builtin_prefetch(a0p + k0 + 32, 0, 3);
      __builtin_prefetch(a1p + k0 + 32, 0, 3);
    }
#pragma unroll
    for (int s = 0; s < 4; ++s) {
      bf16x16 bs = load_bfrag(bp + (size_t)(s * 16) * K + k0);
      acc[0][s] = __builtin_amdgcn_wmma_f32_16x16x32_bf16(false, a0, false, bs, (short)0, acc[0][s], false, false);
      acc[1][s] = __builtin_amdgcn_wmma_f32_16x16x32_bf16(false, a1, false, bs, (short)0, acc[1][s], false, false);
    }
  }

  // C/D layout: VGPR i -> row = i + (lane>>4)*8, col = lane&15
#pragma unroll
  for (int r = 0; r < 2; ++r)
#pragma unroll
    for (int s = 0; s < 4; ++s)
#pragma unroll
      for (int i = 0; i < 8; ++i) {
        const int row = mt * 32 + r * 16 + hw * 8 + i;
        const int col = nb * 64 + s * 16 + m;
        store_out(out + (size_t)row * N + col, acc[r][s][i]);
      }
}

// ---------------------------------------------------------------------------
// RoPE in place on a (T, D)=(T, NH*HD) bf16 buffer, per the reference:
// theta_d = t * base^(-(d%64)/64), pairs (2i, 2i+1) rotated.
// ---------------------------------------------------------------------------
__global__ __launch_bounds__(256) void rope_kernel(bf16_t* __restrict__ qk) {
  const int idx = blockIdx.x * 256 + threadIdx.x;   // T * D/2 threads
  const int t = idx >> 10;                          // D/2 = 1024
  const int r = idx & 1023;
  const int h = r >> 6;                             // HD/2 = 64
  const int i = r & 63;
  const int d0 = 2 * i;
  const size_t base = (size_t)t * D_ + h * HD_ + d0;
  const float x0 = bf16_to_f32(qk[base]);
  const float x1 = bf16_to_f32(qk[base + 1]);
  const int j0 = d0 & 63;
  const int j1 = (d0 + 1) & 63;
  const float th0 = (float)t * powf(10000.0f, -(float)j0 * (1.0f / 64.0f));
  const float th1 = (float)t * powf(10000.0f, -(float)j1 * (1.0f / 64.0f));
  const float o0 = x0 * cosf(th0) - x1 * sinf(th0);
  const float o1 = x1 * cosf(th1) + x0 * sinf(th1);
  qk[base]     = to_bf16(o0);
  qk[base + 1] = to_bf16(o1);
}

// V (T, D) -> V^T (D, T) so P*V B-fragments are contiguous in the key index.
__global__ __launch_bounds__(256) void transpose_kernel(
    const bf16_t* __restrict__ v, bf16_t* __restrict__ vt) {
  const int idx = blockIdx.x * 256 + threadIdx.x;   // T*D threads
  const int t = idx >> 11;                          // D = 2048
  const int d = idx & 2047;
  vt[(size_t)d * T_ + t] = v[idx];
}

// stage one 32-key block of K (32x128) and V^T (128x32) into LDS buffers
static __device__ __forceinline__ void stage_kv(
    const bf16_t* __restrict__ k, const bf16_t* __restrict__ vt,
    bf16_t* ksb, bf16_t* vsb, int kb, int h,
    int kkey, int kseg, int vdim, int vhalf) {
  const bf16_t* gk = k + (size_t)(kb * 32 + kkey) * D_ + h * HD_ + kseg * 16;
  bf16_t* lk = ksb + kkey * 136 + kseg * 16;
  copy_b128(gk, lk);
  copy_b128(gk + 8, lk + 8);
  const bf16_t* gv = vt + (size_t)(h * HD_ + vdim) * T_ + kb * 32 + vhalf * 16;
  bf16_t* lv = vsb + vdim * 40 + vhalf * 16;
  copy_b128(gv, lv);
  copy_b128(gv + 8, lv + 8);
}

// ---------------------------------------------------------------------------
// Flash attention with workgroup-level K/V reuse + double-buffered LDS stage.
// One block (8 waves) = 8 consecutive 16-row query tiles of one head.
// Per 32-key block: S = Q K^T (8 WMMAs), online softmax over 32 columns,
// P (16x32) through LDS into A-layout, O += P V (8 WMMAs); block kb+1 is
// prefetched (async-to-LDS when available) while computing block kb.
// All waves run the uniform key-block count; the causal mask is per element,
// so fully-masked blocks contribute p=0 / alpha=1 and barriers stay uniform.
// ---------------------------------------------------------------------------
__global__ __launch_bounds__(256) void attn_kernel(
    const bf16_t* __restrict__ q, const bf16_t* __restrict__ k,
    const bf16_t* __restrict__ vt, bf16_t* __restrict__ y) {
  __shared__ bf16_t ks[2][32 * 136];   // [buf][key][dim], stride 136
  __shared__ bf16_t vs[2][128 * 40];   // [buf][dim][key], stride 40
  __shared__ bf16_t ps[8 * 16 * 40];   // per-wave 16x32 P tile, stride 40

  const int tid  = threadIdx.x;
  const int lane = tid & 31;
  const int wave = tid >> 5;
  const int h    = blockIdx.x >> 5;    // 32 query groups per head
  const int qb   = blockIdx.x & 31;
  const int qt   = qb * 8 + wave;      // this wave's 16-row query tile
  const int m    = lane & 15;
  const int hw   = lane >> 4;

  // Q fragments for all 4 K-chunks of head_dim (reused across every key block)
  bf16x16 aq[4];
  {
    const bf16_t* qp = q + (size_t)(qt * 16 + m) * D_ + h * HD_;
#pragma unroll
    for (int c = 0; c < 4; ++c) aq[c] = load_afrag(qp + c * 32, hw);
  }

  v8f o[8];
  float mr[8], lr[8];
#pragma unroll
  for (int i = 0; i < 8; ++i) {
    o[i] = v8f{0,0,0,0,0,0,0,0};
    mr[i] = -__builtin_inff();
    lr[i] = 0.0f;
  }

  bf16_t* pw = ps + wave * (16 * 40);

  // cooperative-load work split (256 threads)
  const int kkey  = tid >> 3;          // 0..31  key row
  const int kseg  = tid & 7;           // 0..7   16-dim segment
  const int vdim  = tid >> 1;          // 0..127 dim row
  const int vhalf = tid & 1;           // 0..1   16-key segment

  const int nkb = 4 * (qb + 1);        // key blocks up to the group's max row

  // prologue: stage block 0 into buffer 0
  stage_kv(k, vt, ks[0], vs[0], 0, h, kkey, kseg, vdim, vhalf);
  async_wait_all();
  __syncthreads();

  for (int kb = 0; kb < nkb; ++kb) {
    const int cur = kb & 1;
    // prefetch next key block into the other buffer while we compute
    if (kb + 1 < nkb)
      stage_kv(k, vt, ks[cur ^ 1], vs[cur ^ 1], kb + 1, h, kkey, kseg, vdim, vhalf);

    // ---- S = Q K^T : two 16x16 tiles over 32 keys ----
    v8f s0 = v8f{0,0,0,0,0,0,0,0};
    v8f s1 = v8f{0,0,0,0,0,0,0,0};
#pragma unroll
    for (int c = 0; c < 4; ++c) {
      bf16x16 b0 = load_bfrag(ks[cur] + m * 136 + c * 32 + hw * 16);
      bf16x16 b1 = load_bfrag(ks[cur] + (16 + m) * 136 + c * 32 + hw * 16);
      s0 = __builtin_amdgcn_wmma_f32_16x16x32_bf16(false, aq[c], false, b0, (short)0, s0, false, false);
      s1 = __builtin_amdgcn_wmma_f32_16x16x32_bf16(false, aq[c], false, b1, (short)0, s1, false, false);
    }

    // ---- online softmax across 32 columns ----
    float p0[8], p1[8];
#pragma unroll
    for (int i = 0; i < 8; ++i) {
      const int qrow = qt * 16 + hw * 8 + i;
      float sv0 = s0[i] * SCALE_;
      float sv1 = s1[i] * SCALE_;
      if (kb * 32 + m      > qrow) sv0 = -__builtin_inff();
      if (kb * 32 + 16 + m > qrow) sv1 = -__builtin_inff();
      float rmax = fmaxf(sv0, sv1);
      rmax = fmaxf(rmax, __shfl_xor(rmax, 1));
      rmax = fmaxf(rmax, __shfl_xor(rmax, 2));
      rmax = fmaxf(rmax, __shfl_xor(rmax, 4));
      rmax = fmaxf(rmax, __shfl_xor(rmax, 8));
      const float mnew  = fmaxf(mr[i], rmax);
      const float alpha = expf(mr[i] - mnew);
      p0[i] = expf(sv0 - mnew);
      p1[i] = expf(sv1 - mnew);
      float rsum = p0[i] + p1[i];
      rsum += __shfl_xor(rsum, 1);
      rsum += __shfl_xor(rsum, 2);
      rsum += __shfl_xor(rsum, 4);
      rsum += __shfl_xor(rsum, 8);
      lr[i] = lr[i] * alpha + rsum;
      mr[i] = mnew;
#pragma unroll
      for (int dt = 0; dt < 8; ++dt) o[dt][i] *= alpha;
    }

    // ---- C-layout P (16x32) -> A-layout via per-wave LDS tile ----
#pragma unroll
    for (int i = 0; i < 8; ++i) {
      const int row = hw * 8 + i;
      pw[row * 40 + m]      = to_bf16(p0[i]);
      pw[row * 40 + 16 + m] = to_bf16(p1[i]);
    }
    asm volatile("s_wait_dscnt 0" ::: "memory");
    bf16x16 ap = load_afrag(pw + m * 40, hw);

    // ---- O += P * V over full K=32 ----
#pragma unroll
    for (int dt = 0; dt < 8; ++dt) {
      bf16x16 bv = load_bfrag(vs[cur] + (dt * 16 + m) * 40 + hw * 16);
      o[dt] = __builtin_amdgcn_wmma_f32_16x16x32_bf16(false, ap, false, bv, (short)0, o[dt], false, false);
    }

    async_wait_all();   // next buffer fully landed in LDS
    __syncthreads();    // everyone done reading cur; next data visible
  }

  // ---- epilogue: normalize and write y (T, D) in bf16 for the Wo GEMM ----
#pragma unroll
  for (int dt = 0; dt < 8; ++dt) {
#pragma unroll
    for (int i = 0; i < 8; ++i) {
      const int row = qt * 16 + hw * 8 + i;
      const int col = h * HD_ + dt * 16 + m;
      y[(size_t)row * D_ + col] = to_bf16(o[dt][i] / lr[i]);
    }
  }
}

// ---------------------------------------------------------------------------
extern "C" void kernel_launch(void* const* d_in, const int* in_sizes, int n_in,
                              void* d_out, int out_size, void* d_ws, size_t ws_size,
                              hipStream_t stream) {
  (void)in_sizes; (void)n_in; (void)out_size; (void)ws_size;
  const float* x  = (const float*)d_in[0];
  const float* Wq = (const float*)d_in[1];
  const float* Wk = (const float*)d_in[2];
  const float* Wv = (const float*)d_in[3];
  const float* Wo = (const float*)d_in[4];
  float* out = (float*)d_out;

  char* ws = (char*)d_ws;
  const size_t TD2 = (size_t)T_ * D_ * sizeof(bf16_t);   // 16 MiB
  const size_t DD2 = (size_t)D_ * D_ * sizeof(bf16_t);   // 8 MiB

  bf16_t* x_bf  = (bf16_t*)(ws);                          // also reused as y_bf
  bf16_t* wq_bf = (bf16_t*)(ws + TD2);
  bf16_t* wk_bf = (bf16_t*)(ws + TD2 + DD2);
  bf16_t* wv_bf = (bf16_t*)(ws + TD2 + 2 * DD2);
  bf16_t* wo_bf = (bf16_t*)(ws + TD2 + 3 * DD2);
  bf16_t* q_bf  = (bf16_t*)(ws + TD2 + 4 * DD2);
  bf16_t* k_bf  = (bf16_t*)(ws + 2 * TD2 + 4 * DD2);
  bf16_t* v_bf  = (bf16_t*)(ws + 3 * TD2 + 4 * DD2);
  bf16_t* v_t   = (bf16_t*)(ws + 4 * TD2 + 4 * DD2);
  bf16_t* y_bf  = x_bf;                                   // x dead after projections

  // ---- one-shot fp32 -> bf16 conversion of x and the four weights ----
  cvt_bf16_kernel<<<(T_ * D_ / 8) / 256, 256, 0, stream>>>(x,  x_bf);
  cvt_bf16_kernel<<<(D_ * D_ / 8) / 256, 256, 0, stream>>>(Wq, wq_bf);
  cvt_bf16_kernel<<<(D_ * D_ / 8) / 256, 256, 0, stream>>>(Wk, wk_bf);
  cvt_bf16_kernel<<<(D_ * D_ / 8) / 256, 256, 0, stream>>>(Wv, wv_bf);
  cvt_bf16_kernel<<<(D_ * D_ / 8) / 256, 256, 0, stream>>>(Wo, wo_bf);

  // ---- projections: 32x64 tile per wave -> (M/32)*(N/64)/8 blocks = 512 ----
  const int gemm_blocks = (T_ / 32) * (D_ / 64) / 8;
  gemm_bt_bf16_kernel<bf16_t><<<gemm_blocks, 256, 0, stream>>>(x_bf, wq_bf, q_bf, T_, D_, D_);
  gemm_bt_bf16_kernel<bf16_t><<<gemm_blocks, 256, 0, stream>>>(x_bf, wk_bf, k_bf, T_, D_, D_);
  gemm_bt_bf16_kernel<bf16_t><<<gemm_blocks, 256, 0, stream>>>(x_bf, wv_bf, v_bf, T_, D_, D_);

  // ---- RoPE on q,k ; transpose v -> v^T (D, T) ----
  rope_kernel<<<(T_ * D_ / 2) / 256, 256, 0, stream>>>(q_bf);
  rope_kernel<<<(T_ * D_ / 2) / 256, 256, 0, stream>>>(k_bf);
  transpose_kernel<<<(T_ * D_) / 256, 256, 0, stream>>>(v_bf, v_t);

  // ---- flash attention: 16 heads x 32 query groups = 512 blocks (8 waves) ----
  attn_kernel<<<NH_ * (T_ / 16 / 8), 256, 0, stream>>>(q_bf, k_bf, v_t, y_bf);

  // ---- output projection: y (bf16) x Wo^T -> fp32 out ----
  gemm_bt_bf16_kernel<float><<<gemm_blocks, 256, 0, stream>>>(y_bf, wo_bf, out, T_, D_, D_);
}
